// AutoCorrelation_57775900066271
// MI455X (gfx1250) — matrix operand order repaired and checked by reading, not compiled, not run on hardware
//
#include <hip/hip_runtime.h>
#include <hip/hip_bf16.h>

// AutoCorrelation (Autoformer) for B=16, S=2048, D=512, fp32 in/out.
// corr[tau] = sum_s Q[(s+tau)%S] * K[s]  computed per (b,d) channel with
// bf16 WMMA (16x16x32, f32 accum), then top-7 lags, then weighted circular
// gather of V.  One 256-thread workgroup (8 wave32) per channel.

#define SEQ   2048
#define DCH   512
#define NB    16
#define TOPK  7

typedef __attribute__((ext_vector_type(16))) __bf16 v16bf;
typedef __attribute__((ext_vector_type(8)))  float  v8f;

union Frag16 {
    v16bf        v;
    unsigned int u[8];
};

__device__ __forceinline__ unsigned short f32_to_bf16_rne(float f) {
    unsigned int x = __float_as_uint(f);
    x += 0x7FFFu + ((x >> 16) & 1u);          // round-to-nearest-even
    return (unsigned short)(x >> 16);
}

__global__ __launch_bounds__(256)
void autocorr_wmma_kernel(const float* __restrict__ Q,
                          const float* __restrict__ K,
                          const float* __restrict__ V,
                          float* __restrict__ Out) {
    // ---- LDS staging -------------------------------------------------------
    __shared__ unsigned short Qd [2 * SEQ];        // Q channel, duplicated x2
    __shared__ unsigned short Qd1[2 * SEQ];        // Qd shifted by 1 (odd-base loads)
    __shared__ unsigned short Kd [SEQ + 256 + 32]; // K rotated by +256
    __shared__ float          corr[SEQ];
    __shared__ float          Vl  [SEQ];
    __shared__ float          redv[256];
    __shared__ int            redi[256];
    __shared__ float          wts [TOPK];
    __shared__ int            dls [TOPK];

    const int tid  = threadIdx.x;
    const int chan = blockIdx.x;            // 0 .. B*D-1
    const int b    = chan >> 9;             // / DCH
    const int d    = chan & (DCH - 1);

    // ---- load channel (stride DCH in global; L2 absorbs across blocks) ----
    for (int i = tid; i < SEQ; i += 256) {
        const long base = ((long)b * SEQ + i) * DCH + d;
        const unsigned short qh = f32_to_bf16_rne(Q[base]);
        const unsigned short kh = f32_to_bf16_rne(K[base]);
        Qd[i]        = qh;
        Qd[i + SEQ]  = qh;
        Kd[i + 256]  = kh;                   // Kd[j] = K[(j-256) mod SEQ]
        if (i >= SEQ - 256) Kd[i - (SEQ - 256)] = kh;
        if (i < 32)         Kd[i + SEQ + 256]   = kh;
        Vl[i] = V[base];
    }
    __syncthreads();

    // shifted copy: Qd1[j] = Qd[j+1]  (Qd1[4095] wraps to Q[0])
    for (int j = tid; j < 2 * SEQ; j += 256)
        Qd1[j] = Qd[(j + 1) & (2 * SEQ - 1)];
    __syncthreads();

    // ---- correlation via WMMA ---------------------------------------------
    // wave w owns lag tile [256w, 256w+256): D[m,n] = corr[Lb + m + 16n]
    //   A[m,k] = Qd[k0 + Lb + m + Kidx]   (Hankel slice of Q)
    //   B[k,n] = Kd[256 + k0 + Kidx - 16n]
    {
        const int lane  = tid & 31;
        const int wave  = tid >> 5;
        const int Lb    = wave << 8;
        const int m     = lane & 15;
        const int hi    = lane >> 4;              // 0: low K half, 1: high K half
        const int khalf = hi << 3;                // A half offset (0 or 8)
        const int n     = lane & 15;

        // A base (element index); parity is fixed per lane (= m & 1)
        const int aoff0  = Lb + m + khalf;
        const unsigned int* aptr =
            (const unsigned int*)((m & 1) ? (const unsigned short*)Qd1
                                          : (const unsigned short*)Qd);
        int abase = (aoff0 - (m & 1)) >> 1;       // dword index, advances 16/iter

        // B base (element index 256 + 8*hi - 16*n + k0), always even
        int bbase = (256 + (hi << 3) - (n << 4)) >> 1;
        const unsigned int* bptr = (const unsigned int*)Kd;

        v8f acc = {0.f, 0.f, 0.f, 0.f, 0.f, 0.f, 0.f, 0.f};

        for (int k0 = 0; k0 < SEQ; k0 += 32) {
            Frag16 Af, Bf;
            // A: elements 0..7 <- K = khalf+0..7 ; 8..15 <- K = 16+khalf+0..7
#pragma unroll
            for (int j = 0; j < 4; ++j) {
                Af.u[j]     = aptr[abase + j];
                Af.u[4 + j] = aptr[abase + 8 + j];
            }
            // B: elements 0..7 <- K = 8*hi+0..7 ; 8..15 <- K = 16+8*hi+0..7
#pragma unroll
            for (int j = 0; j < 4; ++j) {
                Bf.u[j]     = bptr[bbase + j];
                Bf.u[4 + j] = bptr[bbase + 8 + j];
            }

            acc = __builtin_amdgcn_wmma_f32_16x16x32_bf16(
                      false, Af.v, false, Bf.v, (short)0, acc, false, false);

            abase += 16;   // 32 elements
            bbase += 16;
        }

        // scatter tile: lane,l vgpr r -> D[M = r + 8*hi, N = lane&15]
#pragma unroll
        for (int r = 0; r < 8; ++r) {
            const int lag = Lb + (r + (hi << 3)) + (n << 4);
            corr[lag] = acc[r];
        }
    }
    __syncthreads();

    // ---- top-7 lags (7 x parallel argmax) ---------------------------------
    for (int it = 0; it < TOPK; ++it) {
        float best = -__builtin_inff();
        int   bidx = 0;
        for (int i = tid; i < SEQ; i += 256) {
            const float v = corr[i];
            if (v > best) { best = v; bidx = i; }
        }
        redv[tid] = best;
        redi[tid] = bidx;
        __syncthreads();
        for (int off = 128; off > 0; off >>= 1) {
            if (tid < off) {
                if (redv[tid + off] > redv[tid]) {
                    redv[tid] = redv[tid + off];
                    redi[tid] = redi[tid + off];
                }
            }
            __syncthreads();
        }
        if (tid == 0) {
            wts[it] = redv[0];
            dls[it] = redi[0];
            corr[redi[0]] = -__builtin_inff();   // exclude for next pass
        }
        __syncthreads();
    }

    // ---- time-delay aggregation: out[t] = sum_i w_i * V[(t+d_i)%S] --------
    float w[TOPK];
    int   dl[TOPK];
#pragma unroll
    for (int i = 0; i < TOPK; ++i) { w[i] = wts[i]; dl[i] = dls[i]; }

    for (int t = tid; t < SEQ; t += 256) {
        float s = 0.f;
#pragma unroll
        for (int i = 0; i < TOPK; ++i)
            s += w[i] * Vl[(t + dl[i]) & (SEQ - 1)];
        Out[((long)b * SEQ + t) * DCH + d] = s;
    }
}

extern "C" void kernel_launch(void* const* d_in, const int* in_sizes, int n_in,
                              void* d_out, int out_size, void* d_ws, size_t ws_size,
                              hipStream_t stream) {
    (void)in_sizes; (void)n_in; (void)d_ws; (void)ws_size; (void)out_size;
    const float* Q = (const float*)d_in[0];
    const float* K = (const float*)d_in[1];
    const float* V = (const float*)d_in[2];
    float* Out = (float*)d_out;

    dim3 grid(NB * DCH);   // 8192 workgroups: one per (b,d) channel
    dim3 block(256);       // 8 wave32
    autocorr_wmma_kernel<<<grid, block, 0, stream>>>(Q, K, V, Out);
}
